// LinearAttention_13709535609201
// MI455X (gfx1250) — matrix operand order repaired and checked by reading, not compiled, not run on hardware
//
#include <hip/hip_runtime.h>

#define BATCH 4
#define CDIM  128
#define NTOK  32768
#define NTILE 512          // 32768 / 64 column-tiles per batch
#define NB1   128          // K1 blocks per batch (grid-stride over 4 tiles each)
#define LN_EPS 1e-5f

typedef __attribute__((ext_vector_type(16))) __bf16 bf16x16;
typedef __attribute__((ext_vector_type(8)))  __bf16 bf16x8;
typedef __attribute__((ext_vector_type(8)))  float  f32x8;
typedef __attribute__((ext_vector_type(4)))  float  f32x4;

#if defined(__has_builtin)
#if __has_builtin(__builtin_amdgcn_global_load_async_to_lds_b128)
#define HAVE_ASYNC_LDS 1
#endif
#if __has_builtin(__builtin_amdgcn_s_wait_asynccnt)
#define HAVE_WAIT_ASYNC 1
#endif
#endif

typedef __attribute__((address_space(3))) void lds_void_t;
typedef __attribute__((address_space(1))) void glb_void_t;
// the async-LDS builtins take pointers to 16-byte int vectors (vector_size spelling)
typedef int i32x4_vs __attribute__((vector_size(16)));
typedef __attribute__((address_space(1))) i32x4_vs g_i32x4;
typedef __attribute__((address_space(3))) i32x4_vs l_i32x4;

#if defined(HAVE_ASYNC_LDS)
__device__ __forceinline__ void async_copy_b128(const void* g, void* l) {
  __builtin_amdgcn_global_load_async_to_lds_b128(
      (g_i32x4*)(glb_void_t*)(void*)g, (l_i32x4*)(lds_void_t*)l, 0, 0);
}
#endif

__device__ __forceinline__ void wait_asynccnt0() {
#if defined(HAVE_WAIT_ASYNC)
  __builtin_amdgcn_s_wait_asynccnt(0);
#else
  asm volatile("s_wait_asynccnt 0x0" ::: "memory");
#endif
}

__device__ __forceinline__ f32x8 wmma_bf16(bf16x16 a, bf16x16 b, f32x8 c) {
  // D = A(16x32 bf16) x B(32x16 bf16) + C(f32)
  return __builtin_amdgcn_wmma_f32_16x16x32_bf16(false, a, false, b, (short)0, c, false, false);
}

__device__ __forceinline__ f32x8 zero8() {
  f32x8 z;
#pragma unroll
  for (int i = 0; i < 8; ++i) z[i] = 0.0f;
  return z;
}

// A-fragment (16x32, M=rowBase+r, K chunk kc) from row-major f32 [rows][ld].
// Lane-half pattern: half0 -> K {0..7,16..23}, half1 -> K {8..15,24..31} (within kc*32).
__device__ __forceinline__ bf16x16 load_a_frag_g(const float* __restrict__ w, int ld,
                                                 int rowBase, int kc, int r, int half) {
  const float* p = w + (size_t)(rowBase + r) * ld + kc * 32 + half * 8;
  f32x4 p0 = *(const f32x4*)(p);
  f32x4 p1 = *(const f32x4*)(p + 4);
  f32x4 p2 = *(const f32x4*)(p + 16);
  f32x4 p3 = *(const f32x4*)(p + 20);
  bf16x16 a;
#pragma unroll
  for (int i = 0; i < 4; ++i) {
    a[i]      = (__bf16)p0[i];
    a[4 + i]  = (__bf16)p1[i];
    a[8 + i]  = (__bf16)p2[i];
    a[12 + i] = (__bf16)p3[i];
  }
  return a;
}

// A-fragment from row-major bf16 LDS [row][K] (stride multiple of 8 elems).
__device__ __forceinline__ bf16x16 load_a_frag_lds(const __bf16* base, int stride,
                                                   int rowBase, int kc, int r, int half) {
  const __bf16* p = base + (size_t)(rowBase + r) * stride + kc * 32 + half * 8;
  bf16x8 lo = *(const bf16x8*)(p);
  bf16x8 hi = *(const bf16x8*)(p + 16);
  return __builtin_shufflevector(lo, hi, 0,1,2,3,4,5,6,7,8,9,10,11,12,13,14,15);
}

// B-fragment (32x16, N=colBase+r, K chunk kc) from LDS stored [n][K] (K contiguous).
// Lane-half pattern: half0 -> K {0..7,16..23}, half1 -> K {8..15,24..31} (within kc*32).
__device__ __forceinline__ bf16x16 load_b_frag_lds(const __bf16* base, int stride,
                                                   int colBase, int kc, int r, int half) {
  const __bf16* p = base + (size_t)(colBase + r) * stride + kc * 32 + half * 8;
  bf16x8 lo = *(const bf16x8*)(p);
  bf16x8 hi = *(const bf16x8*)(p + 16);
  return __builtin_shufflevector(lo, hi, 0,1,2,3,4,5,6,7,8,9,10,11,12,13,14,15);
}

// ---------------------------------------------------------------------------
// K1: qkv GEMM + q d-softmax (-> bf16 ws, layout [b][n][ch]) +
//     WMMA partial exp(k)·v^T and exp(k) row sums
// grid (NB1, BATCH), 256 threads (8 waves)
// ---------------------------------------------------------------------------
__global__ __launch_bounds__(256) void k1_qkv(
    const float* __restrict__ x, const float* __restrict__ wqkv,
    unsigned short* __restrict__ qws_raw, float* __restrict__ ekv_part,
    float* __restrict__ esum_part, int nb) {
  extern __shared__ f32x4 smem4[];
  char* smem = (char*)smem4;
  __bf16* lds_x = (__bf16*)smem;              // [64][136] bf16  (x tile, [n][c])
  float*  q_s   = (float*)(smem + 17408);     // [64][132] f32   (q staging, [n][ch])
  __bf16* k_s   = (__bf16*)(smem + 51200);    // [128][72] bf16  (exp(k), [ch][n])
  __bf16* v_s   = (__bf16*)(smem + 69632);    // [128][72] bf16  (v, [ch][n])
  __bf16* q_w   = reinterpret_cast<__bf16*>(qws_raw);

  const int tid  = threadIdx.x;
  const int wave = tid >> 5, lane = tid & 31, r = lane & 15, half = lane >> 4;
  const int b = blockIdx.y, bx = blockIdx.x;
  const int colw = tid & 63, cw0 = tid >> 6;

  // W_qkv A-fragments held in registers: wave owns row-tiles 3w..3w+2 (rows 48w..48w+47)
  bf16x16 aw[3][4];
#pragma unroll
  for (int i = 0; i < 3; ++i)
#pragma unroll
    for (int kc = 0; kc < 4; ++kc)
      aw[i][kc] = load_a_frag_g(wqkv, CDIM, (wave * 3 + i) * 16, kc, r, half);

  f32x8 acc_kv[2] = {zero8(), zero8()};  // context partial: head wave>>1, d-tile wave&1
  float esum_acc = 0.0f;                 // threads<128 own (h = tid>>5, d = tid&31)

  const float* xb = x + (size_t)b * CDIM * NTOK;

  for (int tile = bx; tile < NTILE; tile += nb) {
    const int n0 = tile * 64;
    // stage x tile -> LDS as bf16 [n][c]
    for (int c = cw0; c < CDIM; c += 4)
      lds_x[colw * 136 + c] = (__bf16)xb[(size_t)c * NTOK + n0 + colw];
    if (tile + nb < NTILE)
      __builtin_prefetch(&xb[(size_t)cw0 * NTOK + (tile + nb) * 64 + colw], 0, 3);
    __syncthreads();

    // qkv GEMM: 24 row-tiles x 4 col-subtiles, K=128
#pragma unroll
    for (int ct = 0; ct < 4; ++ct) {
      f32x8 acc[3] = {zero8(), zero8(), zero8()};
#pragma unroll
      for (int kc = 0; kc < 4; ++kc) {
        bf16x16 bf = load_b_frag_lds(lds_x, 136, ct * 16, kc, r, half);
#pragma unroll
        for (int i = 0; i < 3; ++i) acc[i] = wmma_bf16(aw[i][kc], bf, acc[i]);
      }
      const int col = ct * 16 + r;
#pragma unroll
      for (int i = 0; i < 3; ++i) {
        const int tIdx = wave * 3 + i;  // wave-uniform tile id: q=0..7, k=8..15, v=16..23
        if (tIdx < 8) {                             // q rows -> f32 staging
          float* qp = &q_s[col * 132 + tIdx * 16 + half * 8];
#pragma unroll
          for (int v = 0; v < 8; ++v) qp[v] = acc[i][v];
        } else if (tIdx < 16) {                     // k rows -> exp -> bf16 [ch][n]
          const int ch = (tIdx - 8) * 16 + half * 8;
#pragma unroll
          for (int v = 0; v < 8; ++v) k_s[(ch + v) * 72 + col] = (__bf16)__expf(acc[i][v]);
        } else {                                    // v rows -> bf16 [ch][n]
          const int ch = (tIdx - 16) * 16 + half * 8;
#pragma unroll
          for (int v = 0; v < 8; ++v) v_s[(ch + v) * 72 + col] = (__bf16)acc[i][v];
        }
      }
    }
    __syncthreads();

    // context partial: acc_kv += exp_k(16d x 64n) x v^T(64n x 16e), per head 2 waves
    {
      const int h = wave >> 1, dt = wave & 1;
      const __bf16* ka = k_s + (h * 32 + dt * 16) * 72;
      const __bf16* vb = v_s + (h * 32) * 72;
#pragma unroll
      for (int kc = 0; kc < 2; ++kc) {
        bf16x16 af = load_a_frag_lds(ka, 72, 0, kc, r, half);
#pragma unroll
        for (int et = 0; et < 2; ++et) {
          bf16x16 bf = load_b_frag_lds(vb, 72, et * 16, kc, r, half);
          acc_kv[et] = wmma_bf16(af, bf, acc_kv[et]);
        }
      }
    }
    // q softmax over d (thread = (head, column)), scale by 1/sqrt(32)
    {
      const int h2 = tid >> 6, c2 = tid & 63;
      float* qp = &q_s[c2 * 132 + h2 * 32];
      float m = qp[0];
      for (int d = 1; d < 32; ++d) m = fmaxf(m, qp[d]);
      float s = 0.0f;
      for (int d = 0; d < 32; ++d) { float e = __expf(qp[d] - m); qp[d] = e; s += e; }
      const float inv = 0.17677669529663687f / s;
      for (int d = 0; d < 32; ++d) qp[d] *= inv;
    }
    // exp(k) row-sum partial, accumulated in registers across tiles
    if (tid < 128) {
      const __bf16* kp = k_s + tid * 72;
      float s = 0.0f;
      for (int n = 0; n < 64; ++n) s += (float)kp[n];
      esum_acc += s;
    }
    __syncthreads();

    // coalesced bf16 write of softmaxed q to workspace, layout [b][n][ch]
    {
      const int ch2 = tid & 127, ngrp = tid >> 7;
      for (int nl = ngrp; nl < 64; nl += 2)
        q_w[((size_t)b * NTOK + n0 + nl) * CDIM + ch2] = (__bf16)q_s[nl * 132 + ch2];
    }
    __syncthreads();
  }

  // deterministic per-block partials
  {
    const int h = wave >> 1, dt = wave & 1;
#pragma unroll
    for (int et = 0; et < 2; ++et)
#pragma unroll
      for (int v = 0; v < 8; ++v) {
        const int d = dt * 16 + half * 8 + v;
        const int e = et * 16 + r;
        ekv_part[(((size_t)(b * nb + bx) * 4 + h) * 32 + d) * 32 + e] = acc_kv[et][v];
      }
  }
  if (tid < 128) esum_part[(size_t)(b * nb + bx) * 128 + tid] = esum_acc;
}

// ---------------------------------------------------------------------------
// K2: reduce partials, normalize context, fold w_out: M[c][(h,d)] = sum_e w_out[c,he]*ctx
// grid (BATCH), 256 threads
// ---------------------------------------------------------------------------
__global__ __launch_bounds__(256) void k2_ctx(
    const float* __restrict__ ekv_part, const float* __restrict__ esum_part,
    const float* __restrict__ w_out, float* __restrict__ Mf, int nb) {
  __shared__ float ctx[4096];    // [h][d][e]
  __shared__ float esum[128];    // [h][d]
  const int b = blockIdx.x, tid = threadIdx.x;
  for (int j = tid; j < 4096; j += 256) {
    float s = 0.0f;
    const float* p = ekv_part + (size_t)b * nb * 4096 + j;
    for (int pi = 0; pi < nb; ++pi) s += p[(size_t)pi * 4096];
    ctx[j] = s;
  }
  if (tid < 128) {
    float s = 0.0f;
    const float* p = esum_part + (size_t)b * nb * 128 + tid;
    for (int pi = 0; pi < nb; ++pi) s += p[(size_t)pi * 128];
    esum[tid] = s;
  }
  __syncthreads();
  for (int j = tid; j < CDIM * CDIM; j += 256) {
    const int c = j >> 7, hd = j & 127;
    const int h = hd >> 5, d = hd & 31;
    const float* wrow = w_out + c * CDIM + h * 32;
    const float* crow = ctx + (h * 32 + d) * 32;
    float s = 0.0f;
#pragma unroll
    for (int e = 0; e < 32; ++e) s += wrow[e] * crow[e];
    Mf[(size_t)b * CDIM * CDIM + j] = s / esum[hd];
  }
}

// ---------------------------------------------------------------------------
// K3: out = M[b](128x128) x q(128 x n) + b_out, then LayerNorm over channels.
// q tile pulled into LDS with GLOBAL_LOAD_ASYNC_TO_LDS_B128 when available.
// grid (512, BATCH), 256 threads (8 waves, wave = row-tile)
// ---------------------------------------------------------------------------
__global__ __launch_bounds__(256) void k3_out(
    const unsigned short* __restrict__ qws_raw, const float* __restrict__ Mf,
    const float* __restrict__ b_out, const float* __restrict__ g_ln,
    const float* __restrict__ b_ln, float* __restrict__ out) {
  __shared__ __bf16 q_t[64 * 136];   // [n][ch]
  __shared__ float out_s[64 * 132];  // [n][c]
  __shared__ float red0[256], red1[256];
  __shared__ float stat_m[64], stat_r[64];
  const __bf16* q_w = reinterpret_cast<const __bf16*>(qws_raw);

  const int tid = threadIdx.x, wave = tid >> 5, lane = tid & 31, r = lane & 15, half = lane >> 4;
  const int b = blockIdx.y, n0 = blockIdx.x * 64;
  const int colw = tid & 63, cw0 = tid >> 6;

  // kick off the async LDS fill first so it overlaps the A-fragment setup
#if defined(HAVE_ASYNC_LDS)
#pragma unroll
  for (int it = 0; it < 4; ++it) {
    const int t = tid + it * 256;
    const int nl = t >> 4, ck = (t & 15) * 8;  // 16B chunk of one token's channels
    const __bf16* gp = q_w + ((size_t)b * NTOK + n0 + nl) * CDIM + ck;
    async_copy_b128(gp, &q_t[nl * 136 + ck]);
  }
#else
  {
    const int ch2 = tid & 127, ngrp = tid >> 7;
    for (int nl = ngrp; nl < 64; nl += 2)
      q_t[nl * 136 + ch2] = q_w[((size_t)b * NTOK + n0 + nl) * CDIM + ch2];
  }
#endif

  bf16x16 am[4];
#pragma unroll
  for (int kc = 0; kc < 4; ++kc)
    am[kc] = load_a_frag_g(Mf + (size_t)b * CDIM * CDIM, CDIM, wave * 16, kc, r, half);
  float bias[8];
#pragma unroll
  for (int v = 0; v < 8; ++v) bias[v] = b_out[wave * 16 + half * 8 + v];

#if defined(HAVE_ASYNC_LDS)
  wait_asynccnt0();
#endif
  __syncthreads();

  const size_t gb = (size_t)b * CDIM * NTOK + (size_t)n0 + colw;  // [b][c][n] for output

#pragma unroll
  for (int ct = 0; ct < 4; ++ct) {
    f32x8 acc = zero8();
#pragma unroll
    for (int kc = 0; kc < 4; ++kc) {
      bf16x16 bf = load_b_frag_lds(q_t, 136, ct * 16, kc, r, half);
      acc = wmma_bf16(am[kc], bf, acc);
    }
    const int cc = ct * 16 + r;
    float* op = &out_s[cc * 132 + wave * 16 + half * 8];
#pragma unroll
    for (int v = 0; v < 8; ++v) op[v] = acc[v] + bias[v];
  }
  __syncthreads();

  // LayerNorm stats: 4 partial threads per column
  {
    const int part = tid >> 6;
    const float* p = &out_s[colw * 132 + part * 32];
    float s = 0.0f, s2 = 0.0f;
    for (int c = 0; c < 32; ++c) { const float xv = p[c]; s += xv; s2 += xv * xv; }
    red0[part * 64 + colw] = s;
    red1[part * 64 + colw] = s2;
  }
  __syncthreads();
  if (tid < 64) {
    const float s  = red0[tid] + red0[64 + tid] + red0[128 + tid] + red0[192 + tid];
    const float s2 = red1[tid] + red1[64 + tid] + red1[128 + tid] + red1[192 + tid];
    const float mean = s * (1.0f / 128.0f);
    const float var  = s2 * (1.0f / 128.0f) - mean * mean;
    stat_m[tid] = mean;
    stat_r[tid] = rsqrtf(var + LN_EPS);
  }
  __syncthreads();
  {
    const float mean = stat_m[colw], rstd = stat_r[colw];
    for (int c = cw0; c < CDIM; c += 4)
      out[gb + (size_t)c * NTOK] =
          (out_s[colw * 132 + c] - mean) * rstd * g_ln[c] + b_ln[c];
  }
}

// ---------------------------------------------------------------------------
extern "C" void kernel_launch(void* const* d_in, const int* in_sizes, int n_in,
                              void* d_out, int out_size, void* d_ws, size_t ws_size,
                              hipStream_t stream) {
  (void)in_sizes; (void)n_in; (void)out_size; (void)ws_size;
  const float* inp   = (const float*)d_in[0];
  const float* w_qkv = (const float*)d_in[1];
  const float* w_out = (const float*)d_in[2];
  const float* b_out = (const float*)d_in[3];
  const float* g_ln  = (const float*)d_in[4];
  const float* b_ln  = (const float*)d_in[5];
  float* out = (float*)d_out;
  char* ws = (char*)d_ws;

  // workspace layout (bytes)
  unsigned short* q_ws = (unsigned short*)ws;                       // 4*32768*128 bf16 = 33,554,432
  float* ekv_part  = (float*)(ws + 33554432);                       // 4*NB1*4096 f32   =  8,388,608
  float* esum_part = (float*)(ws + 33554432 + 8388608);             // 4*NB1*128 f32    =    262,144
  float* Mf        = (float*)(ws + 33554432 + 8388608 + 262144);    // 4*128*128 f32    =    262,144

  const size_t shmem1 = 17408 + 33792 + 18432 + 18432;  // 88,064 B dynamic LDS
  k1_qkv<<<dim3(NB1, BATCH), 256, shmem1, stream>>>(inp, w_qkv, q_ws, ekv_part, esum_part, NB1);
  k2_ctx<<<dim3(BATCH), 256, 0, stream>>>(ekv_part, esum_part, w_out, Mf, NB1);
  k3_out<<<dim3(NTILE, BATCH), 256, 0, stream>>>(q_ws, Mf, b_out, g_ln, b_ln, out);
}
